// BktModel_72765335929488
// MI455X (gfx1250) — compile-verified, matrix operand order checked by reading, not run on hardware
//
#include <hip/hip_runtime.h>
#include <hip/hip_bf16.h>

typedef __attribute__((ext_vector_type(16))) _Float16 v16h;
typedef __attribute__((ext_vector_type(8)))  float    v8f;

// ---------------------------------------------------------------------------
// helpers
// ---------------------------------------------------------------------------
__device__ __forceinline__ float softplusf(float x) {
    // stable softplus: max(x,0) + log1p(exp(-|x|))
    return fmaxf(x, 0.0f) + log1pf(__expf(-fabsf(x)));
}
__device__ __forceinline__ float logsigf(float x) {   // log(sigmoid(x))
    return -softplusf(-x);
}
__device__ __forceinline__ float wsum32(float x) {    // wave32 butterfly sum
    x += __shfl_xor(x, 16, 32);
    x += __shfl_xor(x, 8, 32);
    x += __shfl_xor(x, 4, 32);
    x += __shfl_xor(x, 2, 32);
    x += __shfl_xor(x, 1, 32);
    return x;
}

// ---------------------------------------------------------------------------
// Kernel 0: per-KC precompute.
//   tlp[kc][k], k=0..5 = log_sigmoid of (-pL, pL, pF, -pF, -pI0, pI0)
//   TS[kc][v]  = sum over 4 chunk-internal transitions of tlp[trans_ind]
//              = TLP(16x4) @ Count(4x32)  -> v_wmma_f32_16x16x32_f16
// Count entries are integers 0..4 (exact in f16); TLP is split hi/lo in f16
// so the two accumulated WMMAs reproduce the f32 product to ~1e-7.
// One wave per 16-KC tile. EXEC is all ones at every WMMA.
// ---------------------------------------------------------------------------
__global__ void bkt_precompute(const float* __restrict__ dyn,
                               float* __restrict__ tlp_ws,
                               float* __restrict__ ts_ws,
                               int n_kcs) {
    const int lane = threadIdx.x & 31;
    const int tile = blockIdx.x;
    const int mrow = lane & 15;
    const int row  = tile * 16 + mrow;
    const int rowc = (row < n_kcs) ? row : 0;

    const float pL = dyn[rowc * 3 + 0];
    const float pF = dyn[rowc * 3 + 1];
    const float pI = dyn[rowc * 3 + 2];
    const float t0 = logsigf(-pL);   // trans_ind 0 : 0->0
    const float t1 = logsigf( pL);   // trans_ind 1 : 0->1
    const float t2 = logsigf( pF);   // trans_ind 2 : 1->0
    const float t3 = logsigf(-pF);   // trans_ind 3 : 1->1
    const float t4 = logsigf(-pI);   // init h0=0
    const float t5 = logsigf( pI);   // init h0=1

    if (lane < 16 && row < n_kcs) {
        tlp_ws[row * 8 + 0] = t0;
        tlp_ws[row * 8 + 1] = t1;
        tlp_ws[row * 8 + 2] = t2;
        tlp_ws[row * 8 + 3] = t3;
        tlp_ws[row * 8 + 4] = t4;
        tlp_ws[row * 8 + 5] = t5;
    }

    // A (16x32 f16): lanes 0-15 element e carries K=e for e=0..7 (we use K=0..3),
    // lanes 16-31 carry K=8..15 / 24..31 -> zero (K padded with zeros).
    v16h ahi = {};
    v16h alo = {};
    if (lane < 16) {
        const float tk[4] = { t0, t1, t2, t3 };
#pragma unroll
        for (int k = 0; k < 4; ++k) {
            _Float16 h = (_Float16)tk[k];
            ahi[k] = h;
            alo[k] = (_Float16)(tk[k] - (float)h);
        }
    }

    // B (32x16 f16): lanes 0-15 element e carries K=e, column n=lane.
    // Count[k][v] = #{ j in 1..4 : 2*bit(v,5-j) + bit(v,4-j) == k }
    v16h bLo = {};   // trajectory columns v = 0..15
    v16h bHi = {};   // trajectory columns v = 16..31
    if (lane < 16) {
        const int v0 = lane;
        const int v1 = lane + 16;
#pragma unroll
        for (int k = 0; k < 4; ++k) {
            int c0 = 0, c1 = 0;
#pragma unroll
            for (int j = 1; j < 5; ++j) {
                const int i0 = (((v0 >> (5 - j)) & 1) << 1) | ((v0 >> (4 - j)) & 1);
                const int i1 = (((v1 >> (5 - j)) & 1) << 1) | ((v1 >> (4 - j)) & 1);
                c0 += (i0 == k);
                c1 += (i1 == k);
            }
            bLo[k] = (_Float16)c0;
            bHi[k] = (_Float16)c1;
        }
    }

    v8f d0 = {};
    d0 = __builtin_amdgcn_wmma_f32_16x16x32_f16(false, ahi, false, bLo, (short)0, d0, false, false);
    d0 = __builtin_amdgcn_wmma_f32_16x16x32_f16(false, alo, false, bLo, (short)0, d0, false, false);
    v8f d1 = {};
    d1 = __builtin_amdgcn_wmma_f32_16x16x32_f16(false, ahi, false, bHi, (short)0, d1, false, false);
    d1 = __builtin_amdgcn_wmma_f32_16x16x32_f16(false, alo, false, bHi, (short)0, d1, false, false);

    // D layout: lane l<16 VGPR r -> D[m=r,   n=l];
    //           lane l>=16 VGPR r -> D[m=8+r, n=l-16].
    const int ncol  = lane & 15;
    const int mbase = (lane < 16) ? 0 : 8;
#pragma unroll
    for (int r = 0; r < 8; ++r) {
        const int orow = tile * 16 + mbase + r;
        if (orow < n_kcs) {
            ts_ws[orow * 32 + ncol]      = d0[r];
            ts_ws[orow * 32 + 16 + ncol] = d1[r];
        }
    }
}

// ---------------------------------------------------------------------------
// Kernel 1: the fastBKT scan. One wave32 per student, lane = trajectory
// (V = 2^5 = 32 exactly fills a CDNA5 wave). Carry (nh1,nh0) renormalized by
// its max each chunk; the shift cancels in the final log_softmax.
// ---------------------------------------------------------------------------
__global__ __launch_bounds__(256) void bkt_main(
    const float* __restrict__ corr,
    const int*   __restrict__ kc,
    const int*   __restrict__ problem,
    const float* __restrict__ obs_kc,
    const float* __restrict__ obs_prob,
    const float* __restrict__ tlp_ws,
    const float* __restrict__ ts_ws,
    float*       __restrict__ out,
    int B, int T) {

    const int lane = threadIdx.x & 31;
    const int b    = blockIdx.x * (blockDim.x >> 5) + (threadIdx.x >> 5);
    if (b >= B) return;

    const int C   = T / 5;
    const int kcb = kc[b];

    const float okc0 = obs_kc[kcb * 2 + 0];   // guess-logit part
    const float okc1 = obs_kc[kcb * 2 + 1];   // slip-logit part

    const float t0 = tlp_ws[kcb * 8 + 0];
    const float t1 = tlp_ws[kcb * 8 + 1];
    const float t2 = tlp_ws[kcb * 8 + 2];
    const float t3 = tlp_ws[kcb * 8 + 3];
    const float t4 = tlp_ws[kcb * 8 + 4];
    const float t5 = tlp_ws[kcb * 8 + 5];

    const float TSv  = ts_ws[kcb * 32 + lane];          // trans_sum[v=lane]
    const float plp1 = (lane & 1) ? t3 : t1;            // log P(h'=1 | last bit)
    const float plp0 = (lane & 1) ? t2 : t0;            // log P(h'=0 | last bit)
    const bool  bit0 = ((lane >> 4) & 1) != 0;          // first bit of trajectory

    float nh1r = t5;   // chunk 0: init0 with h0=1
    float nh0r = t4;   // chunk 0: init0 with h0=0

    const float* corr_b = corr    + (size_t)b * T;
    const int*   prob_b = problem + (size_t)b * T;
    float*       out_b  = out     + (size_t)b * T * 2;

    for (int c = 0; c < C; ++c) {
        const int tb = c * 5;

        // async-ish warmup of next chunk (global_prefetch_b8)
        if (lane < 5 && c + 1 < C) {
            __builtin_prefetch(&prob_b[tb + 5 + lane], 0, 0);
            __builtin_prefetch(&corr_b[tb + 5 + lane], 0, 0);
        }

        // wave-uniform per-step observation log-likelihood tables
        float lc1[5], lc0[5], li1[5], li0[5], cj[5];
#pragma unroll
        for (int j = 0; j < 5; ++j) {
            cj[j] = corr_b[tb + j];
            const int p  = prob_b[tb + j];
            const float lg = okc0 + obs_prob[p * 2 + 0];
            const float ls = okc1 + obs_prob[p * 2 + 1];
            lc1[j] = -softplusf( ls);   // log P(correct   | h=1) = log sigmoid(-ls)
            lc0[j] = -softplusf(-lg);   // log P(correct   | h=0) = log sigmoid( lg)
            li1[j] = -softplusf(-ls);   // log P(incorrect | h=1) = log sigmoid( ls)
            li0[j] = -softplusf( lg);   // log P(incorrect | h=0) = log sigmoid(-lg)
        }

        const float hl = TSv + (bit0 ? nh1r : nh0r);    // logprob_h (shifted)
        float past = 0.0f;
        float ec[5], ei[5];
#pragma unroll
        for (int j = 0; j < 5; ++j) {
            const bool bj   = ((lane >> (4 - j)) & 1) != 0;
            const float olc = bj ? lc1[j] : lc0[j];
            const float oli = bj ? li1[j] : li0[j];
            const float base = hl + past;
            ec[j] = __expf(base + olc);
            ei[j] = __expf(base + oli);
            past += (cj[j] != 0.0f) ? olc : oli;        // observed step ll
        }
        const float e1 = __expf(plp1 + hl + past);      // seq_ll + pred one
        const float e0 = __expf(plp0 + hl + past);      // seq_ll + pred zero

        // 12 cross-trajectory reductions (logsumexp without shift: terms bounded)
        const float Sc0 = wsum32(ec[0]);
        const float Sc1 = wsum32(ec[1]);
        const float Sc2 = wsum32(ec[2]);
        const float Sc3 = wsum32(ec[3]);
        const float Sc4 = wsum32(ec[4]);
        const float Si0 = wsum32(ei[0]);
        const float Si1 = wsum32(ei[1]);
        const float Si2 = wsum32(ei[2]);
        const float Si3 = wsum32(ei[3]);
        const float Si4 = wsum32(ei[4]);
        const float N1  = wsum32(e1);
        const float N0  = wsum32(e0);

        // lanes 0..4 emit log_softmax([incorrect, correct]) for step j=lane
        if (lane < 5) {
            float sc = Sc0, si = Si0;
            sc = (lane == 1) ? Sc1 : sc;  si = (lane == 1) ? Si1 : si;
            sc = (lane == 2) ? Sc2 : sc;  si = (lane == 2) ? Si2 : si;
            sc = (lane == 3) ? Sc3 : sc;  si = (lane == 3) ? Si3 : si;
            sc = (lane == 4) ? Sc4 : sc;  si = (lane == 4) ? Si4 : si;
            const float lse = __logf(sc + si);
            float2 o;
            o.x = __logf(si) - lse;
            o.y = __logf(sc) - lse;
            *(float2*)&out_b[(tb + lane) * 2] = o;
        }

        // carry update + renormalize (shift cancels in log_softmax)
        const float nh1n = __logf(N1);
        const float nh0n = __logf(N0);
        const float M    = fmaxf(nh1n, nh0n);
        nh1r = nh1n - M;
        nh0r = nh0n - M;
    }
}

// ---------------------------------------------------------------------------
// launch
// ---------------------------------------------------------------------------
extern "C" void kernel_launch(void* const* d_in, const int* in_sizes, int n_in,
                              void* d_out, int out_size, void* d_ws, size_t ws_size,
                              hipStream_t stream) {
    const float* corr     = (const float*)d_in[0];
    const int*   kc       = (const int*)  d_in[1];
    const int*   problem  = (const int*)  d_in[2];
    const float* dyn      = (const float*)d_in[3];
    const float* obs_kc   = (const float*)d_in[4];
    const float* obs_prob = (const float*)d_in[5];
    // d_in[6] = fastbkt_n (structurally 5; hardcoded: V=32 = wave32)

    const int B     = in_sizes[1];
    const int T     = in_sizes[0] / B;
    const int n_kcs = in_sizes[3] / 3;

    float* tlp_ws = (float*)d_ws;                       // n_kcs * 8 floats
    float* ts_ws  = tlp_ws + (size_t)n_kcs * 8;         // n_kcs * 32 floats

    const int tiles = (n_kcs + 15) / 16;
    bkt_precompute<<<tiles, 32, 0, stream>>>(dyn, tlp_ws, ts_ws, n_kcs);

    const int wavesPerBlock = 8;
    const int grid = (B + wavesPerBlock - 1) / wavesPerBlock;
    bkt_main<<<grid, 32 * wavesPerBlock, 0, stream>>>(
        corr, kc, problem, obs_kc, obs_prob, tlp_ws, ts_ws, (float*)d_out, B, T);
}
